// MicroBiMambaBackbone_45956150067255
// MI455X (gfx1250) — compile-verified, matrix operand order checked by reading, not run on hardware
//
#include <hip/hip_runtime.h>
#include <hip/hip_bf16.h>

// Problem constants (match reference)
#define BB      8
#define LL      2048
#define INDIM   5
#define DM      64          // D_MODEL
#define DI      128         // D_INNER
#define NS      16          // N_STATE
#define DTR     4           // DT_RANK
#define CK      4           // conv K
#define NLAY    4           // T = 2*N_LAYERS
#define CH      64          // tokens per chunk
#define NCH     (LL / CH)
#define WXP     48          // W_x padded cols (36 -> 48, multiple of 16)
#define CHX     72          // xi|z row stride: 8-half front pad + 64 tokens (16B aligned)

typedef __attribute__((ext_vector_type(16))) _Float16 v16h;
typedef __attribute__((ext_vector_type(8)))  _Float16 v8h;
typedef __attribute__((ext_vector_type(4)))  _Float16 v4h;
typedef __attribute__((ext_vector_type(8)))  float    v8f;
typedef __attribute__((ext_vector_type(4)))  float    v4f;
typedef __attribute__((ext_vector_type(4)))  unsigned int u32x4;
typedef __attribute__((ext_vector_type(8)))  int      i32x8;
typedef __attribute__((ext_vector_type(4)))  int      i32x4;

// ---------------- LDS layout (bytes, dynamic extern shared; dyn LDS base = offset 0) ----
#define OFF_WIN16   0                                   // 64*256 f16 packed  = 32768
#define OFF_WX16    (OFF_WIN16 + DM*2*DI*2)             // 128*48 f16 packed  = 12288
#define OFF_WOUT16  (OFF_WX16  + DI*WXP*2)              // 128*64 f16 packed  = 16384
#define OFF_WDT     (OFF_WOUT16+ DI*DM*2)               // 4*128 f32
#define OFF_BDT     (OFF_WDT   + DTR*DI*4)
#define OFF_BCONV   (OFF_BDT   + DI*4)
#define OFF_WCONV   (OFF_BCONV + DI*4)
#define OFF_DSKIP   (OFF_WCONV + DI*CK*4)
#define OFF_LNW     (OFF_DSKIP + DI*4)
#define OFF_LNB     (OFF_LNW   + DM*4)
#define OFF_IN      (OFF_LNB   + DM*4)                  // input buf0 (TDM target) 16384
#define OFF_IN2     (OFF_IN    + CH*DM*4)               // input buf1 (TDM target) 16384
#define OFF_LN16    (OFF_IN2   + CH*DM*4)               // 64*64 f16
#define OFF_XIZ16   (OFF_LN16  + CH*DM*2)               // [256][72] f16, pad slots 0..7
#define OFF_XC16    (OFF_XIZ16 + 2*DI*CHX*2)            // 64*128 f16 (row-major [tok][d])
#define OFF_XDBL    (OFF_XC16  + CH*DI*2)               // 64*48 f32  (row-major [tok][48])
#define OFF_DELTA   (OFF_XDBL  + CH*WXP*4)              // 64*128 f32 (reused as out stage)
#define OFF_Y16     (OFF_DELTA + CH*DI*4)               // 64*128 f16 (row-major [tok][d])
#define SMEM_BYTES  (OFF_Y16   + CH*DI*2)               // ~218 KB < 320 KB WGP LDS

// Raw f32 weight staging (preamble only; overlaps chunk scratch, dead at that time)
#define RAW_WIN     OFF_IN                              // 64*256*4 = 65536
#define RAW_WX      (RAW_WIN  + DM*2*DI*4)              // 128*36*4 = 18432
#define RAW_WOUT    (RAW_WX   + DI*36*4)                // 128*64*4 = 32768

__device__ __forceinline__ float silu_f(float x) {
  return x * __builtin_amdgcn_rcpf(1.0f + __expf(-x));   // v_rcp_f32, no IEEE div fixup
}
__device__ __forceinline__ float softplus_f(float x) {
  return (x > 20.0f) ? x : __logf(1.0f + __expf(x));
}

// ---------- WMMA 16x16x32 f16 fragment helpers (CDNA5 ISA 7.12.2 layouts) ----------
__device__ __forceinline__ v16h wmma_load_a(const _Float16* __restrict__ src, int ld) {
  const int lane = threadIdx.x & 31;
  const _Float16* p = src + (lane & 15) * ld + (((lane >> 4) << 3));
  const v8h lo = *(const v8h*)p;          // k = kh .. kh+7
  const v8h hi = *(const v8h*)(p + 16);   // k = kh+16 .. kh+23
  return __builtin_shufflevector(lo, hi, 0,1,2,3,4,5,6,7,8,9,10,11,12,13,14,15);
}
__device__ __forceinline__ v16h wmma_load_bpack(const _Float16* __restrict__ base,
                                                int nt, int kt, int KT) {
  const int lane = threadIdx.x & 31;
  const _Float16* p = base + ((((nt * KT + kt) << 5) + lane) << 4);
  const v8h lo = *(const v8h*)p;
  const v8h hi = *(const v8h*)(p + 8);
  return __builtin_shufflevector(lo, hi, 0,1,2,3,4,5,6,7,8,9,10,11,12,13,14,15);
}
// Stage a KxN f32 weight into fragment-native packed f16 (Ncols = valid source cols).
__device__ __forceinline__ void pack_b(_Float16* __restrict__ dst, const float* __restrict__ src,
                                       int K, int N, int Ncols, int tid) {
  const int KT = K >> 5;
  for (int idx = tid; idx < K * N; idx += 256) {
    const int i    = idx & 15;
    const int lane = (idx >> 4) & 31;
    const int kt   = (idx >> 9) % KT;
    const int nt   = (idx >> 9) / KT;
    const int k = (kt << 5) + (i & 7) + ((i >> 3) << 4) + ((lane >> 4) << 3);
    const int n = (nt << 4) + (lane & 15);
    dst[idx] = (n < Ncols) ? (_Float16)src[k * Ncols + n] : (_Float16)0.0f;
  }
}

// ---------------- TDM: 1D global -> LDS DMA (nElems f32), per ISA 08 D# layout ------
#if __has_builtin(__builtin_amdgcn_tensor_load_to_lds)
#define HAVE_TDM 1
__device__ __forceinline__ void tdm_load_1d(const float* __restrict__ gsrc,
                                            unsigned lds_off, unsigned nElems) {
  const unsigned long long ga = (unsigned long long)(uintptr_t)gsrc;
  u32x4 g0;
  g0[0] = 1u;                                           // count=1, user descriptor
  g0[1] = lds_off;                                      // lds_addr (bytes)
  g0[2] = (unsigned)(ga & 0xFFFFFFFFull);               // global_addr[31:0]
  g0[3] = (unsigned)((ga >> 32) & 0x1FFFFFFull) | (2u << 30);  // addr[56:32] | type=2
  i32x8 g1;
  g1[0] = (2 << 16);                                    // workgroup_mask=0, data_size=4B
  g1[1] = (int)((nElems & 0xFFFFu) << 16);              // tensor_dim0[15:0]
  g1[2] = (int)((nElems >> 16) | (1u << 16));           // dim0[31:16] | tensor_dim1=1
  g1[3] = (int)(nElems << 16);                          // dim1 hi=0 | tile_dim0 (<=65535)
  g1[4] = 1;                                            // tile_dim1=1, tile_dim2=0
  g1[5] = (int)nElems;                                  // tensor_dim0_stride lo
  g1[6] = 0;
  g1[7] = 0;
  const i32x4 gz = {0, 0, 0, 0};
#if __has_include(<hip/amd_detail/amd_gfx1250_TDM.h>)
  const i32x8 gz8 = {0, 0, 0, 0, 0, 0, 0, 0};
  __builtin_amdgcn_tensor_load_to_lds(g0, g1, gz, gz, gz8, 0);   // clang-23 6-arg form
#else
  __builtin_amdgcn_tensor_load_to_lds(g0, g1, gz, gz, 0);        // ROCm 7.2 5-arg form
#endif
}
#else
#define HAVE_TDM 0
#endif

// ---------------- Kernel 1: embedding + PE, writes forward & reversed streams -------
__global__ void embed_kernel(const float* __restrict__ x, const float* __restrict__ W_emb,
                             const float* __restrict__ b_emb, const float* __restrict__ pe,
                             float* __restrict__ hf, float* __restrict__ hb) {
  const int idx = blockIdx.x * blockDim.x + threadIdx.x;
  if (idx >= BB * LL * DM) return;
  const int dm = idx & (DM - 1);
  const int l  = (idx >> 6) & (LL - 1);
  const int b  = idx >> 17;
  const float* xp = x + ((size_t)b * LL + l) * INDIM;
  float s = b_emb[dm] + pe[l * DM + dm];
#pragma unroll
  for (int k = 0; k < INDIM; ++k) s += xp[k] * W_emb[k * DM + dm];
  hf[idx] = s;
  hb[((size_t)b * LL + (LL - 1 - l)) * DM + dm] = s;   // flipped stream
}

// ---------------- Kernel 2: one full Mamba layer, one block per sequence ------------
__global__ __launch_bounds__(256, 1)
void mamba_layer_kernel(float* __restrict__ hbuf,
                        const float* __restrict__ g_lnw,  const float* __restrict__ g_lnb,
                        const float* __restrict__ g_Win,  const float* __restrict__ g_Wconv,
                        const float* __restrict__ g_bconv,const float* __restrict__ g_Wx,
                        const float* __restrict__ g_Wdt,  const float* __restrict__ g_bdt,
                        const float* __restrict__ g_Alog, const float* __restrict__ g_Dskip,
                        const float* __restrict__ g_Wout) {
  extern __shared__ char smem[];
  _Float16* sWinP  = (_Float16*)(smem + OFF_WIN16);
  _Float16* sWxP   = (_Float16*)(smem + OFF_WX16);
  _Float16* sWoutP = (_Float16*)(smem + OFF_WOUT16);
  float*    sWdt   = (float*)   (smem + OFF_WDT);
  float*    sbdt   = (float*)   (smem + OFF_BDT);
  float*    sbconv = (float*)   (smem + OFF_BCONV);
  float*    swconv = (float*)   (smem + OFF_WCONV);
  float*    sdskip = (float*)   (smem + OFF_DSKIP);
  float*    slnw   = (float*)   (smem + OFF_LNW);
  float*    slnb   = (float*)   (smem + OFF_LNB);
  float*    sinb   = (float*)   (smem + OFF_IN);       // 2 x (64*64) double buffer
  _Float16* sln    = (_Float16*)(smem + OFF_LN16);
  _Float16* sxizT  = (_Float16*)(smem + OFF_XIZ16);    // [256][CHX]; tokens at slots 8..71
  _Float16* sxc16  = (_Float16*)(smem + OFF_XC16);
  float*    sxdbl  = (float*)   (smem + OFF_XDBL);
  float*    sdelta = (float*)   (smem + OFF_DELTA);
  float*    sout   = (float*)   (smem + OFF_DELTA);    // reuse after scan
  _Float16* sy16   = (_Float16*)(smem + OFF_Y16);

  const int t   = threadIdx.x;
  const int wid = t >> 5;
  float* seq = hbuf + (size_t)blockIdx.x * LL * DM;

  // ---- DMA raw f32 weights into dead chunk-scratch via TDM, then swizzle-pack ----
#if HAVE_TDM
  if (t < 32) {
    tdm_load_1d(g_Win,  RAW_WIN,  DM * 2 * DI);
    tdm_load_1d(g_Wx,   RAW_WX,   DI * (DTR + 2 * NS));
    tdm_load_1d(g_Wout, RAW_WOUT, DI * DM);
    __builtin_amdgcn_s_wait_tensorcnt(0);
  }
  __syncthreads();
  pack_b(sWinP,  (const float*)(smem + RAW_WIN),  DM, 2 * DI, 2 * DI, t);
  pack_b(sWxP,   (const float*)(smem + RAW_WX),   DI, WXP,    DTR + 2 * NS, t);
  pack_b(sWoutP, (const float*)(smem + RAW_WOUT), DI, DM,     DM, t);
#else
  pack_b(sWinP,  g_Win,  DM, 2 * DI, 2 * DI, t);
  pack_b(sWxP,   g_Wx,   DI, WXP,    DTR + 2 * NS, t);
  pack_b(sWoutP, g_Wout, DI, DM,     DM, t);
#endif
  for (int i = t; i < DTR * DI; i += 256) sWdt[i]  = g_Wdt[i];
  for (int i = t; i < DI * CK;  i += 256) swconv[i] = g_Wconv[i];
  if (t < DI) { sbdt[t] = g_bdt[t]; sbconv[t] = g_bconv[t]; sdskip[t] = g_Dskip[t]; }
  if (t < DM) { slnw[t] = g_lnw[t]; slnb[t] = g_lnb[t]; }

  // ---- per-thread SSM state: d = t>>1, 8 of the 16 states ----
  const int sd = t >> 1;
  const int nb = (t & 1) << 3;
  float hstate[8], Acoef[8];
#pragma unroll
  for (int j = 0; j < 8; ++j) {
    Acoef[j]  = -__expf(g_Alog[sd * NS + nb + j]);
    hstate[j] = 0.0f;
  }
  __syncthreads();                                  // raw weight regions now dead
  if (t < DI) {                                     // causal left pad (conv tail slots)
    sxizT[t * CHX + 5] = (_Float16)0.0f;
    sxizT[t * CHX + 6] = (_Float16)0.0f;
    sxizT[t * CHX + 7] = (_Float16)0.0f;
  }
#if HAVE_TDM
  if (t < 32) tdm_load_1d(seq, OFF_IN, CH * DM);    // chunk 0 -> buf0
#endif

  for (int ch = 0; ch < NCH; ++ch) {
    float* gin = seq + ch * CH * DM;
    float* sin = sinb + (ch & 1) * (CH * DM);

#if HAVE_TDM
    if (t < 32) __builtin_amdgcn_s_wait_tensorcnt(0);      // chunk ch landed
    __syncthreads();
    if (ch + 1 < NCH && t < 32)                            // overlap DMA of next chunk
      tdm_load_1d(gin + CH * DM, OFF_IN + ((ch + 1) & 1) * (CH * DM * 4), CH * DM);
#else
    for (int i = t; i < (CH * DM) / 4; i += 256)
      *(v4f*)(sin + i * 4) = *(const v4f*)(gin + i * 4);
    __syncthreads();
#endif

    // ---- LayerNorm: 4 lanes per token, shfl reduction, all 8 waves busy ----
    {
      const int tk = t >> 2, q = (t & 3) << 4;
      float m = 0.0f, v = 0.0f;
#pragma unroll
      for (int j = 0; j < 16; ++j) { const float xv = sin[tk * DM + q + j]; m += xv; v += xv * xv; }
      m += __shfl_xor(m, 1, 32); m += __shfl_xor(m, 2, 32);
      v += __shfl_xor(v, 1, 32); v += __shfl_xor(v, 2, 32);
      m *= (1.0f / DM);
      const float rs = rsqrtf(v * (1.0f / DM) - m * m + 1e-5f);
#pragma unroll
      for (int j = 0; j < 16; ++j)
        sln[tk * DM + q + j] = (_Float16)((sin[tk * DM + q + j] - m) * rs * slnw[q + j] + slnb[q + j]);
    }
    __syncthreads();

    // ---- GEMM1: ln(64x64) @ W_in(64x256) -> xi|z (transposed, token slots 8..71) ----
    for (int tile = wid; tile < 64; tile += 8) {
      const int mt = tile >> 4, nt = tile & 15;
      v8f c = {};
#pragma unroll
      for (int kk = 0; kk < 2; ++kk) {
        const v16h a = wmma_load_a(sln + mt * 16 * DM + kk * 32, DM);
        const v16h b = wmma_load_bpack(sWinP, nt, kk, 2);
        c = __builtin_amdgcn_wmma_f32_16x16x32_f16(false, a, false, b, (short)0, c, false, false);
      }
      const int lane = t & 31;
      const int n  = nt * 16 + (lane & 15);
      const int mb = mt * 16 + ((lane >> 4) << 3);
      v8h o;
#pragma unroll
      for (int v = 0; v < 8; ++v) o[v] = (_Float16)c[v];
      *(v8h*)(sxizT + n * CHX + 8 + mb) = o;             // one b128 store per lane
    }
    __syncthreads();

    // ---- branchless depthwise causal conv(K=4)+SiLU: register window, b64 loads ----
    {
      const int d = t & 127, lh = t >> 7;                // half-chunk of 32 tokens
      const _Float16* base = sxizT + d * CHX + 4 + lh * 32;  // halves 4..39 (8B aligned)
      _Float16 w[36];
#pragma unroll
      for (int v = 0; v < 9; ++v) *(v4h*)(w + v * 4) = *(const v4h*)(base + v * 4);
      const float k0 = swconv[d * CK + 0], k1 = swconv[d * CK + 1];
      const float k2 = swconv[d * CK + 2], k3 = swconv[d * CK + 3];
      const float bc = sbconv[d];
#pragma unroll
      for (int j = 0; j < 32; ++j) {
        const float s = bc + (float)w[1 + j] * k0 + (float)w[2 + j] * k1
                           + (float)w[3 + j] * k2 + (float)w[4 + j] * k3;
        sxc16[(lh * 32 + j) * DI + d] = (_Float16)silu_f(s);
      }
    }
    __syncthreads();
    for (int i = t; i < 3 * DI; i += 256) {             // carry conv tail into pad slots
      const int r = i >> 7, d = i & 127;
      sxizT[d * CHX + 5 + r] = sxizT[d * CHX + 8 + (CH - 3) + r];
    }
    __syncthreads();

    // ---- GEMM2: xc(64x128) @ W_x(128x48pad) -> dt|B|C ----
    for (int tile = wid; tile < 12; tile += 8) {
      const int mt = tile / 3, nt = tile % 3;
      v8f c = {};
#pragma unroll
      for (int kk = 0; kk < 4; ++kk) {
        const v16h a = wmma_load_a(sxc16 + mt * 16 * DI + kk * 32, DI);
        const v16h b = wmma_load_bpack(sWxP, nt, kk, 4);
        c = __builtin_amdgcn_wmma_f32_16x16x32_f16(false, a, false, b, (short)0, c, false, false);
      }
      const int lane = t & 31;
      const int n  = nt * 16 + (lane & 15);
      const int mb = mt * 16 + ((lane >> 4) << 3);
#pragma unroll
      for (int v = 0; v < 8; ++v) sxdbl[(mb + v) * WXP + n] = c[v];
    }
    __syncthreads();

    // ---- delta = softplus(dt @ W_dt + b_dt) (K=4, VALU; dt row read as one b128) ----
    for (int i = t; i < CH * DI; i += 256) {
      const int l = i >> 7, d = i & 127;
      const v4f dtv = *(const v4f*)(sxdbl + l * WXP);
      float s = sbdt[d];
#pragma unroll
      for (int r = 0; r < DTR; ++r) s += dtv[r] * sWdt[r * DI + d];
      sdelta[i] = softplus_f(s);
    }
    __syncthreads();

    // ---- selective scan: sequential in l, parallel over 2048 (d,n) states ----
    for (int l = 0; l < CH; ++l) {
      const float dlt = sdelta[l * DI + sd];
      const float xcv = (float)sxc16[l * DI + sd];
      const float dx  = dlt * xcv;
      const v4f B0 = *(const v4f*)(sxdbl + l * WXP + DTR + nb);          // b128
      const v4f B1 = *(const v4f*)(sxdbl + l * WXP + DTR + nb + 4);
      const v4f C0 = *(const v4f*)(sxdbl + l * WXP + DTR + NS + nb);
      const v4f C1 = *(const v4f*)(sxdbl + l * WXP + DTR + NS + nb + 4);
      float acc = 0.0f;
#pragma unroll
      for (int j = 0; j < 4; ++j) {
        hstate[j] = __expf(dlt * Acoef[j]) * hstate[j] + dx * B0[j];
        acc += hstate[j] * C0[j];
      }
#pragma unroll
      for (int j = 0; j < 4; ++j) {
        hstate[4 + j] = __expf(dlt * Acoef[4 + j]) * hstate[4 + j] + dx * B1[j];
        acc += hstate[4 + j] * C1[j];
      }
      acc += __shfl_xor(acc, 1, 32);       // both lanes now hold the full 16-state sum
      const float yv = acc + sdskip[sd] * xcv;
      const float zv = (float)sxizT[(DI + sd) * CHX + 8 + l];
      sy16[l * DI + sd] = (_Float16)(yv * silu_f(zv));   // lane pair writes same value
    }
    __syncthreads();

    // ---- GEMM3: y(64x128) @ W_out(128x64) -> staged in LDS (delta region dead) ----
    for (int tile = wid; tile < 16; tile += 8) {
      const int mt = tile >> 2, nt = tile & 3;
      v8f c = {};
#pragma unroll
      for (int kk = 0; kk < 4; ++kk) {
        const v16h a = wmma_load_a(sy16 + mt * 16 * DI + kk * 32, DI);
        const v16h b = wmma_load_bpack(sWoutP, nt, kk, 4);
        c = __builtin_amdgcn_wmma_f32_16x16x32_f16(false, a, false, b, (short)0, c, false, false);
      }
      const int lane = t & 31;
      const int n  = nt * 16 + (lane & 15);
      const int mb = mt * 16 + ((lane >> 4) << 3);
#pragma unroll
      for (int v = 0; v < 8; ++v) sout[(mb + v) * DM + n] = c[v];
    }
    __syncthreads();

    // ---- residual add + coalesced b128 store to global ----
    for (int i = t; i < (CH * DM) / 4; i += 256) {
      const v4f r = *(const v4f*)(sin + i * 4);
      const v4f y = *(const v4f*)(sout + i * 4);
      *(v4f*)(gin + i * 4) = r + y;
    }
    __syncthreads();
  }
}

// ---------------- Kernel 3: mean-pool both streams (flip-invariant) + head proj -----
__global__ __launch_bounds__(128)
void head_kernel(const float* __restrict__ hf, const float* __restrict__ hb,
                 const float* __restrict__ W_proj, const float* __restrict__ b_proj,
                 float* __restrict__ out) {
  __shared__ float zc[2 * DM];
  const int b = blockIdx.x, t = threadIdx.x;
  const float* src = (t < DM) ? hf : hb;
  const int d = t & (DM - 1);
  const float* p = src + (size_t)b * LL * DM + d;
  float acc = 0.0f;
  for (int l = 0; l < LL; ++l) acc += p[l * DM];
  zc[t] = acc * (1.0f / LL);
  __syncthreads();
  if (t < DM) {
    float s = b_proj[t];
#pragma unroll 8
    for (int j = 0; j < 2 * DM; ++j) s += zc[j] * W_proj[j * DM + t];
    out[b * DM + t] = s;
  }
}

// ------------------------------------------------------------------------------------
extern "C" void kernel_launch(void* const* d_in, const int* in_sizes, int n_in,
                              void* d_out, int out_size, void* d_ws, size_t ws_size,
                              hipStream_t stream) {
  (void)in_sizes; (void)n_in; (void)out_size; (void)ws_size;
  const float* x      = (const float*)d_in[0];
  const float* W_emb  = (const float*)d_in[1];
  const float* b_emb  = (const float*)d_in[2];
  const float* pe     = (const float*)d_in[3];
  const float* ln_w   = (const float*)d_in[4];
  const float* ln_b   = (const float*)d_in[5];
  const float* W_in   = (const float*)d_in[6];
  const float* W_conv = (const float*)d_in[7];
  const float* b_conv = (const float*)d_in[8];
  const float* W_x    = (const float*)d_in[9];
  const float* W_dt   = (const float*)d_in[10];
  const float* b_dt   = (const float*)d_in[11];
  const float* A_log  = (const float*)d_in[12];
  const float* D_skip = (const float*)d_in[13];
  const float* W_out  = (const float*)d_in[14];
  const float* W_proj = (const float*)d_in[15];
  const float* b_proj = (const float*)d_in[16];
  float* out = (float*)d_out;

  float* hf = (float*)d_ws;                       // (B,L,64)
  float* hb = hf + (size_t)BB * LL * DM;          // (B,L,64), reversed stream

  static_assert(SMEM_BYTES < 320 * 1024, "LDS budget");
  hipFuncSetAttribute((const void*)mamba_layer_kernel,
                      hipFuncAttributeMaxDynamicSharedMemorySize, SMEM_BYTES);

  embed_kernel<<<(BB * LL * DM + 255) / 256, 256, 0, stream>>>(x, W_emb, b_emb, pe, hf, hb);

  for (int tl = 0; tl < NLAY; ++tl) {
    float* buf = (tl < 2) ? hf : hb;
    mamba_layer_kernel<<<BB, 256, SMEM_BYTES, stream>>>(
        buf,
        ln_w  + tl * DM,        ln_b  + tl * DM,
        W_in  + tl * DM * 2 * DI,
        W_conv+ tl * DI * CK,   b_conv+ tl * DI,
        W_x   + tl * DI * (DTR + 2 * NS),
        W_dt  + tl * DTR * DI,  b_dt  + tl * DI,
        A_log + tl * DI * NS,   D_skip+ tl * DI,
        W_out + tl * DI * DM);
  }

  head_kernel<<<BB, 128, 0, stream>>>(hf, hb, W_proj, b_proj, out);
}
